// CRF_82918638616928
// MI455X (gfx1250) — compile-verified, matrix operand order checked by reading, not run on hardware
//
#include <hip/hip_runtime.h>

typedef __attribute__((ext_vector_type(16))) _Float16 v16h;
typedef __attribute__((ext_vector_type(8)))  float    v8f;

#define B_ 1024
#define S_ 1024
#define T_ 32

static __device__ __forceinline__ float fmaxd(float a, float b) { return a > b ? a : b; }

#define LOG2E_F 1.4426950408889634f
#define LN2_F   0.6931471805599453f

// fast transcendentals: raw v_exp_f32 / v_log_f32, no denorm guards
static __device__ __forceinline__ float fast_exp(float x) {
    return __builtin_amdgcn_exp2f(x * LOG2E_F);
}
static __device__ __forceinline__ float fast_log(float x) {
    return __builtin_amdgcn_logf(x) * LN2_F;
}

// ---------------------------------------------------------------------------
// Kernel A: precompute B-operand staging for E = exp(trans) in f16.
// K is stored INTERLEAVED to match the LDS A-staging in crf_forward:
//   within a 32-half row, half position p holds k = (p>>1) + (p&1)*16.
// A-fragment half i of lane (h = lane>>4):  p = (i<8) ? h*8+i : 16+h*8+(i-8).
// B element for (tile, lane, i):  N = tile*16 + (lane&15), K = perm(p).
// Stored contiguously: bstage[(tile*32 + lane)*16 + i]  (32B per lane).
// ---------------------------------------------------------------------------
__global__ void crf_prep(const float* __restrict__ trans, _Float16* __restrict__ bstage) {
    int idx = blockIdx.x * blockDim.x + threadIdx.x;   // 0 .. 1023
    if (idx >= 2 * 32 * 16) return;
    int i    = idx & 15;
    int lane = (idx >> 4) & 31;
    int tile = idx >> 9;
    int h    = lane >> 4;
    int n    = tile * 16 + (lane & 15);
    int p    = (i < 8) ? (h * 8 + i) : (16 + h * 8 + (i - 8));
    int k    = (p >> 1) + ((p & 1) ? 16 : 0);
    bstage[idx] = (_Float16)__expf(trans[k * T_ + n]);
}

// ---------------------------------------------------------------------------
// Kernel B: per-batch numerator + labels + packed cond bits
// (one 256-thread block per batch row)
// ---------------------------------------------------------------------------
__global__ void crf_numer(const float* __restrict__ em, const int* __restrict__ tags,
                          const unsigned char* __restrict__ mask,
                          const float* __restrict__ startT, const float* __restrict__ trans,
                          const float* __restrict__ endT,
                          float* __restrict__ numer, float* __restrict__ labels,
                          unsigned* __restrict__ condbits) {
    const int b   = blockIdx.x;
    const int tid = threadIdx.x;   // 256 threads
    __shared__ int   sh_i[256];
    __shared__ float sh_f[256];

    int hmin = S_ - 1, tmax = 0, cnt = 0;
    for (int j = tid; j < S_; j += 256) {
        if (mask[(size_t)b * S_ + j]) {
            hmin = min(hmin, j);
            tmax = max(tmax, j);
            cnt++;
        }
    }
    sh_i[tid] = hmin; __syncthreads();
    for (int s = 128; s > 0; s >>= 1) { if (tid < s) sh_i[tid] = min(sh_i[tid], sh_i[tid + s]); __syncthreads(); }
    const int head = sh_i[0]; __syncthreads();

    sh_i[tid] = tmax; __syncthreads();
    for (int s = 128; s > 0; s >>= 1) { if (tid < s) sh_i[tid] = max(sh_i[tid], sh_i[tid + s]); __syncthreads(); }
    const int tail = sh_i[0]; __syncthreads();

    sh_i[tid] = cnt; __syncthreads();
    for (int s = 128; s > 0; s >>= 1) { if (tid < s) sh_i[tid] += sh_i[tid + s]; __syncthreads(); }
    const int nlab = sh_i[0]; __syncthreads();

    // packed cond bits: cond(j) = (j>=1) && mask[b][j] && (j != head)
    if (tid < S_ / 32) {
        unsigned w = 0;
        for (int jj = 0; jj < 32; jj++) {
            int j = tid * 32 + jj;
            if (j >= 1 && mask[(size_t)b * S_ + j] && j != head) w |= (1u << jj);
        }
        condbits[b * (S_ / 32) + tid] = w;
    }

    float acc = 0.f;
    for (int j = tid; j < S_; j += 256) {
        if (j >= 1 && mask[(size_t)b * S_ + j] && j != head) {
            int tp = tags[(size_t)b * S_ + j - 1];
            int tc = tags[(size_t)b * S_ + j];
            acc += trans[tp * T_ + tc] + em[((size_t)b * S_ + j) * T_ + tc];
        }
    }
    sh_f[tid] = acc; __syncthreads();
    for (int s = 128; s > 0; s >>= 1) { if (tid < s) sh_f[tid] += sh_f[tid + s]; __syncthreads(); }

    if (tid == 0) {
        float num = 0.f;
        if (nlab != 0) {
            int ht = tags[(size_t)b * S_ + head];
            int tt = tags[(size_t)b * S_ + tail];
            num = startT[ht] + em[((size_t)b * S_ + head) * T_ + ht] + sh_f[0] + endT[tt];
        }
        numer[b]  = num;
        labels[b] = (float)nlab;
    }
}

// record head index per row (used by forward init)
__global__ void crf_heads(const unsigned char* __restrict__ mask, int* __restrict__ headArr) {
    const int b   = blockIdx.x;
    const int tid = threadIdx.x;   // 256
    __shared__ int sh_i[256];
    int hmin = S_ - 1;
    for (int j = tid; j < S_; j += 256)
        if (mask[(size_t)b * S_ + j]) hmin = min(hmin, j);
    sh_i[tid] = hmin; __syncthreads();
    for (int s = 128; s > 0; s >>= 1) { if (tid < s) sh_i[tid] = min(sh_i[tid], sh_i[tid + s]); __syncthreads(); }
    if (tid == 0) headArr[b] = sh_i[0];
}

// ---------------------------------------------------------------------------
// Kernel C helpers
// ---------------------------------------------------------------------------
union V16 { v16h v; uint4 u[2]; };
union PK  { _Float16 hh[2]; unsigned u; };

static __device__ __forceinline__ void load_em_step(const float* __restrict__ em,
                                                    int b0, int h, int lr, int j,
                                                    float (&e0)[8], float (&e1)[8],
                                                    bool do_prefetch) {
#pragma unroll
    for (int v = 0; v < 8; v++) {
        int M = v + h * 8;
        size_t base = ((size_t)(b0 + M) * S_ + (size_t)j) * T_;
        e0[v] = em[base + lr];
        e1[v] = em[base + 16 + lr];
        if (do_prefetch) __builtin_prefetch(&em[base + 8 * T_], 0, 1);
    }
}

// one forward step: rowmax -> exp -> LDS transpose -> 2x WMMA -> epilogue
static __device__ __forceinline__ void crf_step(float (&s0)[8], float (&s1)[8],
                                                const float (&e0)[8], const float (&e1)[8],
                                                const unsigned (&cw)[8], int jj,
                                                const v16h& btv0, const v16h& btv1,
                                                unsigned* ldsA32, int h, int lr) {
    // row max over all 32 tags
    float rm[8];
#pragma unroll
    for (int v = 0; v < 8; v++) {
        float m = fmaxd(s0[v], s1[v]);
        m = fmaxd(m, __shfl_xor(m, 1, 32));
        m = fmaxd(m, __shfl_xor(m, 2, 32));
        m = fmaxd(m, __shfl_xor(m, 4, 32));
        m = fmaxd(m, __shfl_xor(m, 8, 32));
        rm[v] = m;
    }

    // a = exp(score - rowmax) -> packed f16 pair -> one b32 store per row
#pragma unroll
    for (int v = 0; v < 8; v++) {
        int M = v + h * 8;
        PK pk;
        pk.hh[0] = (_Float16)fast_exp(s0[v] - rm[v]);   // k = lr
        pk.hh[1] = (_Float16)fast_exp(s1[v] - rm[v]);   // k = 16 + lr
        ldsA32[M * 16 + lr] = pk.u;
    }
    __syncthreads();

    // A operand: lane row = lr, two b128 chunks (K order matches bstage)
    V16 a;
    const uint4* lp = (const uint4*)ldsA32;
    a.u[0] = lp[lr * 4 + h];
    a.u[1] = lp[lr * 4 + 2 + h];
    __syncthreads();   // WAR guard before next step's stores

    v8f d0 = {}, d1 = {};
    d0 = __builtin_amdgcn_wmma_f32_16x16x32_f16(false, a.v, false, btv0, (short)0, d0, false, false);
    d1 = __builtin_amdgcn_wmma_f32_16x16x32_f16(false, a.v, false, btv1, (short)0, d1, false, false);

    // score' = em + rowmax + log(p), gated per-row by cond bit
#pragma unroll
    for (int v = 0; v < 8; v++) {
        bool cond = (cw[v] >> jj) & 1u;
        float n0 = e0[v] + rm[v] + fast_log(d0[v]);
        float n1 = e1[v] + rm[v] + fast_log(d1[v]);
        s0[v] = cond ? n0 : s0[v];
        s1[v] = cond ? n1 : s1[v];
    }
}

// ---------------------------------------------------------------------------
// Kernel C: WMMA forward scan. One wave handles 16 batch rows.
// State in C/D layout: s0[v] = score(M = v + 8*h, N = lr),
//                      s1[v] = score(M = same,    N = 16 + lr).
// Emission loads are software-pipelined one step ahead (ping-pong buffers),
// so the only wait on the serial chain is the 1KB LDS transpose round-trip.
// ---------------------------------------------------------------------------
__global__ __launch_bounds__(32) void crf_forward(
    const float* __restrict__ em,
    const float* __restrict__ startT, const float* __restrict__ endT,
    const _Float16* __restrict__ bstage, const int* __restrict__ headArr,
    const unsigned* __restrict__ condbits,
    const float* __restrict__ labels, float* __restrict__ denom) {

    __shared__ unsigned ldsA32[16 * 16];   // [M][dword d] : {k=d lo16, k=16+d hi16}

    const int lane = threadIdx.x;
    const int h    = lane >> 4;      // 0 -> rows 0..7, 1 -> rows 8..15
    const int lr   = lane & 15;
    const int b0   = blockIdx.x * 16;

    // constant B tiles of E = exp(trans): 32 contiguous bytes per lane
    V16 bt0, bt1;
    const uint4* bs = (const uint4*)bstage;
    bt0.u[0] = bs[(0 * 32 + lane) * 2 + 0];
    bt0.u[1] = bs[(0 * 32 + lane) * 2 + 1];
    bt1.u[0] = bs[(1 * 32 + lane) * 2 + 0];
    bt1.u[1] = bs[(1 * 32 + lane) * 2 + 1];

    int headM[8];
#pragma unroll
    for (int v = 0; v < 8; v++) headM[v] = headArr[b0 + v + h * 8];

    const float st0 = startT[lr], st1 = startT[16 + lr];

    // score0 = start + emissions[b, head]
    float s0[8], s1[8];
#pragma unroll
    for (int v = 0; v < 8; v++) {
        int M = v + h * 8;
        size_t base = ((size_t)(b0 + M) * S_ + (size_t)headM[v]) * T_;
        s0[v] = st0 + em[base + lr];
        s1[v] = st1 + em[base + 16 + lr];
    }

    // pipeline: preload emissions for step j = 0
    float ea0[8], ea1[8], eb0[8], eb1[8];
    load_em_step(em, b0, h, lr, 0, ea0, ea1, true);

    for (int jw = 0; jw < S_ / 32; jw++) {
        // one packed cond word per row covers the next 32 steps
        unsigned cw[8];
#pragma unroll
        for (int v = 0; v < 8; v++) cw[v] = condbits[(b0 + v + h * 8) * (S_ / 32) + jw];

        for (int jj = 0; jj < 32; jj += 2) {
            const int j  = jw * 32 + jj;
            const int j1 = (j + 1 < S_) ? j + 1 : S_ - 1;   // uniform clamp (SALU)
            const int j2 = (j + 2 < S_) ? j + 2 : S_ - 1;

            load_em_step(em, b0, h, lr, j1, eb0, eb1, false);      // issue for j+1
            crf_step(s0, s1, ea0, ea1, cw, jj,     bt0.v, bt1.v, ldsA32, h, lr);

            load_em_step(em, b0, h, lr, j2, ea0, ea1, true);       // issue for j+2
            crf_step(s0, s1, eb0, eb1, cw, jj + 1, bt0.v, bt1.v, ldsA32, h, lr);
        }
    }

    // denominator = logsumexp(score_final + end) per row
    const float en0 = endT[lr], en1 = endT[16 + lr];
#pragma unroll
    for (int v = 0; v < 8; v++) {
        float x0 = s0[v] + en0;
        float x1 = s1[v] + en1;
        float m = fmaxd(x0, x1);
        m = fmaxd(m, __shfl_xor(m, 1, 32));
        m = fmaxd(m, __shfl_xor(m, 2, 32));
        m = fmaxd(m, __shfl_xor(m, 4, 32));
        m = fmaxd(m, __shfl_xor(m, 8, 32));
        float p = fast_exp(x0 - m) + fast_exp(x1 - m);
        p += __shfl_xor(p, 1, 32);
        p += __shfl_xor(p, 2, 32);
        p += __shfl_xor(p, 4, 32);
        p += __shfl_xor(p, 8, 32);
        if (lr == 0) {
            int M = v + h * 8;
            float d = m + fast_log(p);
            denom[b0 + M] = (labels[b0 + M] != 0.f) ? d : 0.f;
        }
    }
}

// ---------------------------------------------------------------------------
// Kernel D: final scalar combine
// ---------------------------------------------------------------------------
__global__ void crf_finish(const float* __restrict__ numer, const float* __restrict__ denom,
                           const float* __restrict__ labels, float* __restrict__ out) {
    __shared__ float sa[256], sc[256];
    const int tid = threadIdx.x;
    float a = 0.f, c = 0.f;
    for (int b = tid; b < B_; b += 256) {
        float l   = labels[b];
        float llh = denom[b] - numer[b];
        a += llh / (l + 1e-6f);
        c += (l != 0.f) ? 1.f : 0.f;
    }
    sa[tid] = a; sc[tid] = c; __syncthreads();
    for (int s = 128; s > 0; s >>= 1) {
        if (tid < s) { sa[tid] += sa[tid + s]; sc[tid] += sc[tid + s]; }
        __syncthreads();
    }
    if (tid == 0) out[0] = sa[0] / (sc[0] + 1e-6f);
}

// ---------------------------------------------------------------------------
// Launch
// ---------------------------------------------------------------------------
extern "C" void kernel_launch(void* const* d_in, const int* in_sizes, int n_in,
                              void* d_out, int out_size, void* d_ws, size_t ws_size,
                              hipStream_t stream) {
    const float*         em     = (const float*)d_in[0];
    const int*           tags   = (const int*)d_in[1];
    const unsigned char* mask   = (const unsigned char*)d_in[2];   // jnp bool -> 1 byte
    const float*         startT = (const float*)d_in[3];
    const float*         trans  = (const float*)d_in[4];
    const float*         endT   = (const float*)d_in[5];

    float*    ws       = (float*)d_ws;
    float*    numer    = ws;                          // [B]
    float*    denom    = ws + B_;                     // [B]
    float*    labels   = ws + 2 * B_;                 // [B]
    int*      headArr  = (int*)(ws + 3 * B_);         // [B]
    _Float16* bstage   = (_Float16*)(ws + 4 * B_);    // [2*32*16] halves (2KB)
    unsigned* condbits = (unsigned*)(ws + 4 * B_ + 512); // [B * S/32] words (128KB)

    crf_prep   <<<4,       256, 0, stream>>>(trans, bstage);
    crf_heads  <<<B_,      256, 0, stream>>>(mask, headArr);
    crf_numer  <<<B_,      256, 0, stream>>>(em, tags, mask, startT, trans, endT,
                                             numer, labels, condbits);
    crf_forward<<<B_ / 16, 32,  0, stream>>>(em, startT, endT, bstage, headArr,
                                             condbits, labels, denom);
    crf_finish <<<1,       256, 0, stream>>>(numer, denom, labels, (float*)d_out);
}
